// QuantiZ_20890720928598
// MI455X (gfx1250) — matrix operand order repaired and checked by reading, not compiled, not run on hardware
//
#include <hip/hip_runtime.h>
#include <hip/hip_bf16.h>

typedef __bf16          v16bf __attribute__((ext_vector_type(16)));
typedef unsigned short  v16u  __attribute__((ext_vector_type(16)));
typedef float           v8f   __attribute__((ext_vector_type(8)));
typedef unsigned int    u32x4 __attribute__((ext_vector_type(4)));
typedef int             i32x4 __attribute__((ext_vector_type(4)));
typedef int             i32x8 __attribute__((ext_vector_type(8)));

// float -> bf16 bits, round-to-nearest-even
__device__ __forceinline__ unsigned short f2bf(float f) {
    unsigned u = __float_as_uint(f);
    u += 0x7FFFu + ((u >> 16) & 1u);
    return (unsigned short)(u >> 16);
}
__device__ __forceinline__ float bf2f(unsigned short h) {
    return __uint_as_float(((unsigned)h) << 16);
}

// ---------------------------------------------------------------------------
// Kernel 1: codebook -> bf16 hi/lo split + 0.5*||e||^2   (one block per code)
// ---------------------------------------------------------------------------
__global__ void vq_prep_kernel(const float* __restrict__ cb,
                               unsigned short* __restrict__ cbHi,
                               unsigned short* __restrict__ cbLo,
                               float* __restrict__ halfEE) {
    __shared__ float red[256];
    const int row = blockIdx.x;
    const int t   = threadIdx.x;
    const size_t o = (size_t)row * 256 + t;
    float f = cb[o];
    unsigned short h = f2bf(f);
    unsigned short l = f2bf(f - bf2f(h));
    cbHi[o] = h;
    cbLo[o] = l;
    red[t] = f * f;
    __syncthreads();
    for (int s = 128; s > 0; s >>= 1) {
        if (t < s) red[t] += red[t + s];
        __syncthreads();
    }
    if (t == 0) halfEE[row] = 0.5f * red[0];
}

// ---------------------------------------------------------------------------
// TDM descriptor: copy a 16-row x 512-byte tile (contiguous in global) into
// LDS with 32B padding after every 512B row (LDS row stride 544B).
// Descriptor layout per CDNA5 ISA.
// ---------------------------------------------------------------------------
__device__ __forceinline__ void tdm_load_tile(const void* gsrc, unsigned lds_byte_off) {
    unsigned long long ga = (unsigned long long)(uintptr_t)gsrc;

    u32x4 g0;
    g0.x = 1u;                                             // count=1, is_restore=0
    g0.y = lds_byte_off;                                   // lds_addr
    g0.z = (unsigned)(ga & 0xFFFFFFFFu);                   // global_addr[31:0]
    g0.w = (unsigned)((ga >> 32) & 0x01FFFFFFu) | (2u << 30); // addr[56:32] | type=2

    i32x8 g1;
    // data_size=0 (1B), pad_enable=1, pad_interval=6 (128 DW = 512B),
    // pad_amount=7 (8 DW = 32B), workgroup_mask=0
    g1[0] = (int)((1u << 20) | (6u << 22) | (7u << 25));
    g1[1] = (int)(512u << 16);        // tensor_dim0[15:0]=512
    g1[2] = (int)(16u << 16);         // tensor_dim0 hi=0 | tensor_dim1[15:0]=16
    g1[3] = (int)(512u << 16);        // tensor_dim1 hi=0 | tile_dim0=512
    g1[4] = 16;                       // tile_dim1=16 | tile_dim2=0
    g1[5] = 512;                      // tensor_dim0_stride[31:0]=512
    g1[6] = 0;
    g1[7] = 0;

    i32x4 gz  = {0, 0, 0, 0};                   // 2-D tensor: groups 2/3 unused
    i32x8 gz8 = {0, 0, 0, 0, 0, 0, 0, 0};       // extra group (clang-23 6-arg form)
    __builtin_amdgcn_tensor_load_to_lds(g0, g1, gz, gz, gz8, 0);
}

// ---------------------------------------------------------------------------
// Kernel 2: WMMA GEMM (scores = z . e - 0.5||e||^2) + running argmax per row.
// Block = 128 threads (4 waves), max-register allocation (no spills).
// Double-buffered TDM staging: tile t+1 DMA overlaps tile t's 24 WMMAs.
// ---------------------------------------------------------------------------
#define LDS_ROW 272   // bf16 elems per LDS row (544B stride = 512B data + 32B pad)

__global__ void __launch_bounds__(128, 1)
vq_argmax_kernel(const float* __restrict__ encode,
                 const unsigned short* __restrict__ cbHi,
                 const unsigned short* __restrict__ cbLo,
                 const float* __restrict__ halfEE,
                 int ncode,
                 float* __restrict__ zidxF,
                 int*   __restrict__ zidxI) {
    __shared__ __align__(32) unsigned short ldsHi[2][16 * LDS_ROW];
    __shared__ __align__(32) unsigned short ldsLo[2][16 * LDS_ROW];
    __shared__ float sVal[64 * 16];
    __shared__ int   sIdx[64 * 16];

    const int tid  = threadIdx.x;
    const int wave = tid >> 5;
    const int lane = tid & 31;
    const int half = lane >> 4;
    const int lm   = lane & 15;

    const size_t mBase = (size_t)blockIdx.x * 64 + wave * 16;

    // ---- Load A rows (fp32 -> bf16 hi/lo) into WMMA A operand, contiguous k order (k = 16*half + i) ----
    v16u aHi[8], aLo[8];
    {
        const float* arow = encode + (mBase + lm) * 256;
#pragma unroll
        for (int c = 0; c < 8; ++c) {
            float4 q[4];
            const float4* p0 = (const float4*)(arow + c * 32 + half * 16);
            q[0] = p0[0]; q[1] = p0[1]; q[2] = p0[2]; q[3] = p0[3];
            const float* tf = (const float*)q;
#pragma unroll
            for (int i = 0; i < 16; ++i) {
                unsigned short h = f2bf(tf[i]);
                aHi[c][i] = h;
                aLo[c][i] = f2bf(tf[i] - bf2f(h));
            }
        }
    }

    const unsigned hiOff[2] = {(unsigned)(uintptr_t)&ldsHi[0][0],
                               (unsigned)(uintptr_t)&ldsHi[1][0]};
    const unsigned loOff[2] = {(unsigned)(uintptr_t)&ldsLo[0][0],
                               (unsigned)(uintptr_t)&ldsLo[1][0]};

    float bestV[8];
    int   bestI[8];
#pragma unroll
    for (int v = 0; v < 8; ++v) { bestV[v] = -3.0e38f; bestI[v] = 0; }

    const int ntiles = ncode >> 4;

    // Prologue: stage tile 0 into buffer 0
    if (wave == 0) {
        tdm_load_tile(cbHi, hiOff[0]);
        tdm_load_tile(cbLo, loOff[0]);
    }

    for (int t = 0; t < ntiles; ++t) {
        const int buf      = t & 1;
        const int codeBase = t << 4;

        __syncthreads();   // all waves consumed buffer buf^1 (tile t-1); safe to overwrite
        if (wave == 0) {
            if (t + 1 < ntiles) {
                // Issue next tile into the other buffer, then wait until only
                // those 2 ops remain outstanding (=> current tile's 2 are done;
                // same-wave tensor ops complete in order).
                tdm_load_tile(cbHi + (size_t)(codeBase + 16) * 256, hiOff[buf ^ 1]);
                tdm_load_tile(cbLo + (size_t)(codeBase + 16) * 256, loOff[buf ^ 1]);
                __builtin_amdgcn_s_wait_tensorcnt(2);
            } else {
                __builtin_amdgcn_s_wait_tensorcnt(0);
            }
        }
        __syncthreads();   // current tile visible to all waves

        const float hee = halfEE[codeBase + lm];

        v8f acc = {0.f, 0.f, 0.f, 0.f, 0.f, 0.f, 0.f, 0.f};
#pragma unroll
        for (int c = 0; c < 8; ++c) {
            const v16u bhu = *(const v16u*)&ldsHi[buf][lm * LDS_ROW + c * 32 + half * 16];
            const v16u blu = *(const v16u*)&ldsLo[buf][lm * LDS_ROW + c * 32 + half * 16];
            v16bf bh = __builtin_bit_cast(v16bf, bhu);
            v16bf bl = __builtin_bit_cast(v16bf, blu);
            v16bf ah = __builtin_bit_cast(v16bf, aHi[c]);
            v16bf al = __builtin_bit_cast(v16bf, aLo[c]);
            acc = __builtin_amdgcn_wmma_f32_16x16x32_bf16(false, al, false, bh,
                                                          (short)0, acc, false, false);
            acc = __builtin_amdgcn_wmma_f32_16x16x32_bf16(false, ah, false, bl,
                                                          (short)0, acc, false, false);
            acc = __builtin_amdgcn_wmma_f32_16x16x32_bf16(false, ah, false, bh,
                                                          (short)0, acc, false, false);
        }

        // C layout: lane l holds col (l&15); VGPR v holds row v + 8*(l>>4)
#pragma unroll
        for (int v = 0; v < 8; ++v) {
            float s = acc[v] - hee;
            if (s > bestV[v]) { bestV[v] = s; bestI[v] = codeBase + lm; }
        }
    }

    // ---- Cross-lane argmax reduction (16 column-classes per row) ----
#pragma unroll
    for (int v = 0; v < 8; ++v) {
        int wgRow = wave * 16 + v + 8 * half;
        sVal[wgRow * 16 + lm] = bestV[v];
        sIdx[wgRow * 16 + lm] = bestI[v];
    }
    __syncthreads();
    if (tid < 64) {
        float bv = sVal[tid * 16];
        int   bi = sIdx[tid * 16];
#pragma unroll
        for (int j = 1; j < 16; ++j) {
            float v2 = sVal[tid * 16 + j];
            if (v2 > bv) { bv = v2; bi = sIdx[tid * 16 + j]; }
        }
        size_t row = (size_t)blockIdx.x * 64 + tid;
        zidxF[row] = (float)bi;
        zidxI[row] = bi;
    }
}

// ---------------------------------------------------------------------------
// Kernel 3: quant = codebook[zidx]  (1KB row copy, 64 threads/row, float4)
// ---------------------------------------------------------------------------
__global__ void vq_gather_kernel(const float* __restrict__ cb,
                                 const int* __restrict__ zidxI,
                                 float* __restrict__ quant) {
    const int tid = threadIdx.x;
    const size_t row = (size_t)blockIdx.x * 4 + (tid >> 6);
    const int q = tid & 63;
    const int idx = zidxI[row];
    ((float4*)quant)[row * 64 + q] = ((const float4*)cb)[(size_t)idx * 64 + q];
}

// ---------------------------------------------------------------------------
extern "C" void kernel_launch(void* const* d_in, const int* in_sizes, int n_in,
                              void* d_out, int out_size, void* d_ws, size_t ws_size,
                              hipStream_t stream) {
    const float* encode   = (const float*)d_in[0];  // (b*n, 256) fp32
    const float* codebook = (const float*)d_in[1];  // (ncode, 256) fp32

    const int    C     = 256;
    const size_t M     = (size_t)in_sizes[0] / C;   // 65536
    const int    ncode = in_sizes[1] / C;           // 4096

    char* ws = (char*)d_ws;
    unsigned short* cbHi   = (unsigned short*)ws;                             // 2 MB
    unsigned short* cbLo   = (unsigned short*)(ws + (size_t)ncode * C * 2);   // 2 MB
    float*          halfEE = (float*)(ws + (size_t)ncode * C * 4);            // 16 KB
    int*            zidxI  = (int*)(ws + (size_t)ncode * C * 4 + (size_t)ncode * 4);

    float* zidxF = (float*)d_out;        // first M floats: indices
    float* quant = (float*)d_out + M;    // then M*256 floats: gathered codes

    vq_prep_kernel<<<ncode, 256, 0, stream>>>(codebook, cbHi, cbLo, halfEE);
    vq_argmax_kernel<<<(unsigned)(M / 64), 128, 0, stream>>>(
        encode, cbHi, cbLo, halfEE, ncode, zidxF, zidxI);
    vq_gather_kernel<<<(unsigned)(M / 4), 256, 0, stream>>>(codebook, zidxI, quant);
}
